// LinearTimeSelfAttention1d_22686017257608
// MI455X (gfx1250) — compile-verified, matrix operand order checked by reading, not run on hardware
//
#include <hip/hip_runtime.h>

// ---------------- types / helpers ----------------
typedef __attribute__((ext_vector_type(16))) __bf16 v16bf;
typedef __attribute__((ext_vector_type(8)))  float  v8f;
typedef unsigned short u16;
typedef unsigned int   u32;

#define B_      16
#define C_      512
#define W_      4096
#define O3_     1536      // 3*HIDDEN
#define HID_    512
#define HEADS_  8
#define HD_     64
#define GROUPS_ 32
#define EPS_    1e-5f

__device__ __forceinline__ u16 f2bf(float f) {          // f32 -> bf16 RNE
  u32 u = __builtin_bit_cast(u32, f);
  u32 r = u + 0x7fffu + ((u >> 16) & 1u);
  return (u16)(r >> 16);
}
__device__ __forceinline__ float bf2f(u16 h) {
  u32 u = ((u32)h) << 16;
  return __builtin_bit_cast(float, u);
}

// One lane's 16-bf16 WMMA fragment: elements 0..7 = K[lk..lk+7],
// elements 8..15 = K[16+lk .. 16+lk+7]  (lk = (lane>>4)*8), matching the
// CDNA5 16-bit A/B VGPR layout.  p must be 16B aligned.
__device__ __forceinline__ v16bf ld_frag(const u16* p) {
  union { uint4 q[2]; v16bf v; } f;
  f.q[0] = *(const uint4*)(p);
  f.q[1] = *(const uint4*)(p + 16);
  return f.v;
}

__device__ __forceinline__ v8f wmma_bf16(v16bf a, v16bf b, v8f c) {
  return __builtin_amdgcn_wmma_f32_16x16x32_bf16(false, a, false, b, (short)0, c,
                                                 false, false);
}

// Async global->LDS DMA (CDNA5, ASYNCcnt-tracked).  Copies 32 bytes:
// INST_OFFSET is added to BOTH the LDS and the global address per the ISA
// so one address
// pair feeds both b128 transfers.
__device__ __forceinline__ void async_copy32(const u16* g, u16* l) {
  asm volatile(
      "global_load_async_to_lds_b128 %0, %1, off\n\t"
      "global_load_async_to_lds_b128 %0, %1, off offset:16"
      :: "v"((u32)(size_t)l), "v"((unsigned long long)(size_t)g)
      : "memory");
}
__device__ __forceinline__ void wait_async0() {
#if __has_builtin(__builtin_amdgcn_s_wait_asynccnt)
  __builtin_amdgcn_s_wait_asynccnt(0);
#else
  asm volatile("s_wait_asynccnt 0x0" ::: "memory");
#endif
}

// ---------------- 1) groupnorm statistics ----------------
// One block per (b,g).  Each group = 16 consecutive channels * W = 65536
// contiguous floats.  Output (mu, rsigma).
__global__ __launch_bounds__(256) void k_gn_stats(const float* __restrict__ x,
                                                  float2* __restrict__ stats) {
  const int bg = blockIdx.x;                       // b*32 + g
  const float4* src = (const float4*)(x + (size_t)bg * (16 * W_));
  float s = 0.f, sq = 0.f;
  for (int i = threadIdx.x; i < (16 * W_) / 4; i += 256) {
    float4 v = src[i];
    s  += v.x + v.y + v.z + v.w;
    sq += v.x * v.x + v.y * v.y + v.z * v.z + v.w * v.w;
  }
  __shared__ float rS[256], rQ[256];
  rS[threadIdx.x] = s; rQ[threadIdx.x] = sq;
  __syncthreads();
  for (int off = 128; off > 0; off >>= 1) {
    if (threadIdx.x < off) {
      rS[threadIdx.x] += rS[threadIdx.x + off];
      rQ[threadIdx.x] += rQ[threadIdx.x + off];
    }
    __syncthreads();
  }
  if (threadIdx.x == 0) {
    const float n  = 16.f * (float)W_;
    float mu  = rS[0] / n;
    float var = rQ[0] / n - mu * mu;
    stats[bg] = make_float2(mu, rsqrtf(var + EPS_));
  }
}

// ---------------- 2) per-(b,c) scale/shift ----------------
__global__ void k_scaleshift(const float2* __restrict__ stats,
                             const float* __restrict__ gw,
                             const float* __restrict__ gb,
                             float2* __restrict__ ss) {
  int i = blockIdx.x * 256 + threadIdx.x;
  if (i >= B_ * C_) return;
  int b = i >> 9, c = i & 511;
  float2 st = stats[b * GROUPS_ + (c >> 4)];
  float sc = st.y * gw[c];
  ss[i] = make_float2(sc, gb[c] - st.x * sc);
}

// ---------------- 3) weights -> bf16 ----------------
__global__ void k_cvt_w(const float* __restrict__ wq, const float* __restrict__ wo,
                        u16* __restrict__ wqb, u16* __restrict__ wob) {
  int i = (blockIdx.x * 256 + threadIdx.x) * 4;
  const int N1 = O3_ * C_;                         // 786432
  if (i < N1) {
    float4 v = *(const float4*)(wq + i);
    wqb[i] = f2bf(v.x); wqb[i + 1] = f2bf(v.y);
    wqb[i + 2] = f2bf(v.z); wqb[i + 3] = f2bf(v.w);
  } else {
    int j = i - N1;
    if (j < HID_ * C_) {
      float4 v = *(const float4*)(wo + j);
      wob[j] = f2bf(v.x); wob[j + 1] = f2bf(v.y);
      wob[j + 2] = f2bf(v.z); wob[j + 3] = f2bf(v.w);
    }
  }
}

// ---------------- 4) QKV GEMM with fused groupnorm ----------------
// qkv[b] (1536 x 4096) = w_qkv_bf (1536 x 512) * xn[b] (512 x 4096) + b_qkv
// xn computed on the fly:  xn = x * scale[c] + shift[c].
// Block tile 128x256, 8 waves (2x4), wave tile 64x64 (16 wmma accum tiles).
// A tile staged with async global->LDS DMA; B tile staged via VALU (f32->bf16
// conversion + transpose into LDS).
__global__ __launch_bounds__(256) void k_qkv_gemm(
    const u16* __restrict__ Aw, const float* __restrict__ x,
    const float2* __restrict__ ss, const float* __restrict__ bq,
    u16* __restrict__ qkv) {
  const int m0 = blockIdx.x * 128, n0 = blockIdx.y * 256, b = blockIdx.z;
  const int tid = threadIdx.x;
  const int wv = tid >> 5, lane = tid & 31;
  const int wm = wv >> 2, wn = wv & 3;
  const int lm = lane & 15, lk = (lane >> 4) * 8;

  __shared__ __align__(16) u16 lA[128 * 40];       // [m][k], padded stride 40
  __shared__ __align__(16) u16 lB[256 * 40];       // [n][k] (transposed)

  const v8f vz = {0.f, 0.f, 0.f, 0.f, 0.f, 0.f, 0.f, 0.f};
  v8f acc[4][4];
#pragma unroll
  for (int i = 0; i < 4; ++i)
#pragma unroll
    for (int j = 0; j < 4; ++j) acc[i][j] = vz;

  const int ar = tid >> 1, ah = (tid & 1) * 16;    // A stage: 128 rows x 2 halves
  const int br = tid >> 3, bc = (tid & 7) * 32;    // B stage: 32 rows x 8 chunks

  for (int kb = 0; kb < C_; kb += 32) {
    // stage A 128x32 bf16: async DMA straight into LDS (2 x b128 per thread)
    async_copy32(Aw + (size_t)(m0 + ar) * C_ + kb + ah, &lA[ar * 40 + ah]);
    { // stage B 32x256: read x f32, apply groupnorm, write transposed bf16
      const int c = kb + br;
      const float2 s = ss[b * C_ + c];
      const float4* g = (const float4*)(x + ((size_t)b * C_ + c) * W_ + n0 + bc);
#pragma unroll
      for (int i = 0; i < 8; ++i) {
        float4 v = g[i];
        u16* d = &lB[(bc + i * 4) * 40 + br];
        d[0]   = f2bf(fmaf(v.x, s.x, s.y));
        d[40]  = f2bf(fmaf(v.y, s.x, s.y));
        d[80]  = f2bf(fmaf(v.z, s.x, s.y));
        d[120] = f2bf(fmaf(v.w, s.x, s.y));
      }
    }
    if (kb + 32 < C_)
      __builtin_prefetch(x + ((size_t)b * C_ + kb + 32 + br) * W_ + n0 + bc, 0, 1);
    wait_async0();
    __syncthreads();

    v16bf bf[4];
#pragma unroll
    for (int nn = 0; nn < 4; ++nn)
      bf[nn] = ld_frag(&lB[(wn * 64 + nn * 16 + lm) * 40 + lk]);
#pragma unroll
    for (int mm = 0; mm < 4; ++mm) {
      v16bf af = ld_frag(&lA[(wm * 64 + mm * 16 + lm) * 40 + lk]);
#pragma unroll
      for (int nn = 0; nn < 4; ++nn) acc[mm][nn] = wmma_bf16(af, bf[nn], acc[mm][nn]);
    }
    __syncthreads();
  }

  const int hi8 = (lane >> 4) * 8;
#pragma unroll
  for (int mm = 0; mm < 4; ++mm)
#pragma unroll
    for (int nn = 0; nn < 4; ++nn)
#pragma unroll
      for (int r = 0; r < 8; ++r) {
        int o = m0 + wm * 64 + mm * 16 + r + hi8;
        int n = n0 + wn * 64 + nn * 16 + lm;
        qkv[((size_t)b * O3_ + o) * W_ + n] = f2bf(acc[mm][nn][r] + bq[o]);
      }
}

// ---------------- 5) softmax over W on the K rows (in place) ----------------
__global__ __launch_bounds__(256) void k_softmax(u16* __restrict__ qkv) {
  const int row = blockIdx.x;                     // B*512 rows
  const int b = row >> 9, rr = row & 511;
  u16* p = qkv + ((size_t)b * O3_ + HID_ + rr) * W_ + threadIdx.x * 16;
  union { uint4 q[2]; u16 h[16]; } u;
  u.q[0] = *(const uint4*)p;
  u.q[1] = *(const uint4*)(p + 8);
  float v[16], mx = -3.4e38f;
#pragma unroll
  for (int i = 0; i < 16; ++i) { v[i] = bf2f(u.h[i]); mx = fmaxf(mx, v[i]); }

  __shared__ float red[256];
  red[threadIdx.x] = mx;
  __syncthreads();
  for (int off = 128; off > 0; off >>= 1) {
    if (threadIdx.x < off)
      red[threadIdx.x] = fmaxf(red[threadIdx.x], red[threadIdx.x + off]);
    __syncthreads();
  }
  mx = red[0];
  __syncthreads();

  float s = 0.f;
#pragma unroll
  for (int i = 0; i < 16; ++i) { v[i] = __expf(v[i] - mx); s += v[i]; }
  red[threadIdx.x] = s;
  __syncthreads();
  for (int off = 128; off > 0; off >>= 1) {
    if (threadIdx.x < off) red[threadIdx.x] += red[threadIdx.x + off];
    __syncthreads();
  }
  const float inv = 1.f / red[0];
#pragma unroll
  for (int i = 0; i < 16; ++i) u.h[i] = f2bf(v[i] * inv);
  *(uint4*)p       = u.q[0];
  *(uint4*)(p + 8) = u.q[1];
}

// ---------------- 6) context[d][e] = sum_n k[d,n] v[e,n]  (split-K) --------
// blockIdx.x = (b,h), blockIdx.y = K-quarter (1024 columns each).  Both
// operands row-contiguous along K(=n): fragments load straight from global
// (global_load_b128), no LDS.  f32 partials -> deterministic reduce kernel.
__global__ __launch_bounds__(128) void k_context(const u16* __restrict__ qkv,
                                                 float* __restrict__ ctxp) {
  const int bh = blockIdx.x, b = bh >> 3, h = bh & 7;
  const int kq = blockIdx.y;                       // 0..3
  const int tid = threadIdx.x, wv = tid >> 5, lane = tid & 31;
  const int lm = lane & 15, lk = (lane >> 4) * 8;
  const u16* Kp = qkv + ((size_t)b * O3_ + HID_ + h * HD_) * W_;
  const u16* Vp = qkv + ((size_t)b * O3_ + 2 * HID_ + h * HD_) * W_;

  const v8f vz = {0.f, 0.f, 0.f, 0.f, 0.f, 0.f, 0.f, 0.f};
  v8f acc[4] = {vz, vz, vz, vz};
  const u16* arow = Kp + (size_t)(wv * 16 + lm) * W_;
  const int k0 = kq * (W_ / 4), k1 = k0 + (W_ / 4);
  for (int k = k0; k < k1; k += 32) {
    v16bf af = ld_frag(arow + k + lk);
#pragma unroll
    for (int nn = 0; nn < 4; ++nn) {
      v16bf bf = ld_frag(Vp + (size_t)(nn * 16 + lm) * W_ + k + lk);
      acc[nn] = wmma_bf16(af, bf, acc[nn]);
    }
  }
  const int hi8 = (lane >> 4) * 8;
  float* out = ctxp + ((size_t)kq * (B_ * HEADS_) + bh) * (HD_ * HD_);
#pragma unroll
  for (int nn = 0; nn < 4; ++nn)
#pragma unroll
    for (int r = 0; r < 8; ++r) {
      int d = wv * 16 + r + hi8, e = nn * 16 + lm;
      out[d * HD_ + e] = acc[nn][r];               // [d][e] partial
    }
}

__global__ void k_ctx_reduce(const float* __restrict__ ctxp,
                             u16* __restrict__ ctx) {
  const int i = blockIdx.x * 256 + threadIdx.x;    // B*HEADS*64*64 = 524288
  const size_t Q = (size_t)B_ * HEADS_ * HD_ * HD_;
  float s = ctxp[i] + ctxp[i + Q] + ctxp[i + 2 * Q] + ctxp[i + 3 * Q];
  ctx[i] = f2bf(s);
}

// ---------------- 7) W2[b][c][h*64+d] = sum_e w_out[c][h*64+e]*ctx[d][e] ----
// Folds the attention-out GEMM into the output projection weights so the
// final pass is one big GEMM.  One block (8 waves) per (b,h); M=512, N=64, K=64.
__global__ __launch_bounds__(256) void k_w2(const u16* __restrict__ wo,
                                            const u16* __restrict__ ctx,
                                            u16* __restrict__ w2) {
  const int bh = blockIdx.x, b = bh >> 3, h = bh & 7;
  const int tid = threadIdx.x, wv = tid >> 5, lane = tid & 31;
  const int lm = lane & 15, lk = (lane >> 4) * 8;

  const v8f vz = {0.f, 0.f, 0.f, 0.f, 0.f, 0.f, 0.f, 0.f};
  v8f acc[4][4];
#pragma unroll
  for (int i = 0; i < 4; ++i)
#pragma unroll
    for (int j = 0; j < 4; ++j) acc[i][j] = vz;

#pragma unroll
  for (int kb = 0; kb < HD_; kb += 32) {
    v16bf bf[4];
#pragma unroll
    for (int nn = 0; nn < 4; ++nn)   // B frag: column d=nn*16+lm, k=e contiguous
      bf[nn] = ld_frag(ctx + ((size_t)bh * HD_ + nn * 16 + lm) * HD_ + kb + lk);
#pragma unroll
    for (int mm = 0; mm < 4; ++mm) {
      int c = wv * 64 + mm * 16 + lm;
      v16bf af = ld_frag(wo + (size_t)c * HID_ + h * HD_ + kb + lk);
#pragma unroll
      for (int nn = 0; nn < 4; ++nn) acc[mm][nn] = wmma_bf16(af, bf[nn], acc[mm][nn]);
    }
  }
  const int hi8 = (lane >> 4) * 8;
#pragma unroll
  for (int mm = 0; mm < 4; ++mm)
#pragma unroll
    for (int nn = 0; nn < 4; ++nn)
#pragma unroll
      for (int r = 0; r < 8; ++r) {
        int c = wv * 64 + mm * 16 + r + hi8, d = nn * 16 + lm;
        w2[((size_t)b * HID_ + c) * HID_ + h * HD_ + d] = f2bf(acc[mm][nn][r]);
      }
}

// ---------------- 8) final: out[b] = W2[b] (512x512) * q[b] (512x4096) + b_out
__global__ __launch_bounds__(256) void k_final_gemm(
    const u16* __restrict__ w2, const u16* __restrict__ qkv,
    const float* __restrict__ bo, float* __restrict__ out) {
  const int m0 = blockIdx.x * 128, n0 = blockIdx.y * 256, b = blockIdx.z;
  const int tid = threadIdx.x;
  const int wv = tid >> 5, lane = tid & 31;
  const int wm = wv >> 2, wn = wv & 3;
  const int lm = lane & 15, lk = (lane >> 4) * 8;
  const u16* A = w2 + (size_t)b * HID_ * HID_;

  __shared__ __align__(16) u16 lA[128 * 40];
  __shared__ __align__(16) u16 lB[256 * 40];

  const v8f vz = {0.f, 0.f, 0.f, 0.f, 0.f, 0.f, 0.f, 0.f};
  v8f acc[4][4];
#pragma unroll
  for (int i = 0; i < 4; ++i)
#pragma unroll
    for (int j = 0; j < 4; ++j) acc[i][j] = vz;

  const int ar = tid >> 1, ah = (tid & 1) * 16;
  const int br = tid >> 3, bc = (tid & 7) * 32;

  for (int kb = 0; kb < HID_; kb += 32) {
    // stage A 128x32 via async global->LDS DMA
    async_copy32(A + (size_t)(m0 + ar) * HID_ + kb + ah, &lA[ar * 40 + ah]);
    { // stage B 32x256 from q rows (bf16), transposed into LDS
      const u16* s = qkv + ((size_t)b * O3_ + kb + br) * W_ + n0 + bc;
      union { uint4 q[4]; u16 h[32]; } u;
      const uint4* g = (const uint4*)s;
      u.q[0] = g[0]; u.q[1] = g[1]; u.q[2] = g[2]; u.q[3] = g[3];
#pragma unroll
      for (int i = 0; i < 32; ++i) lB[(bc + i) * 40 + br] = u.h[i];
    }
    if (kb + 32 < HID_)
      __builtin_prefetch(qkv + ((size_t)b * O3_ + kb + 32 + br) * W_ + n0 + bc, 0, 1);
    wait_async0();
    __syncthreads();

    v16bf bf[4];
#pragma unroll
    for (int nn = 0; nn < 4; ++nn)
      bf[nn] = ld_frag(&lB[(wn * 64 + nn * 16 + lm) * 40 + lk]);
#pragma unroll
    for (int mm = 0; mm < 4; ++mm) {
      v16bf af = ld_frag(&lA[(wm * 64 + mm * 16 + lm) * 40 + lk]);
#pragma unroll
      for (int nn = 0; nn < 4; ++nn) acc[mm][nn] = wmma_bf16(af, bf[nn], acc[mm][nn]);
    }
    __syncthreads();
  }

  const int hi8 = (lane >> 4) * 8;
#pragma unroll
  for (int mm = 0; mm < 4; ++mm)
#pragma unroll
    for (int nn = 0; nn < 4; ++nn)
#pragma unroll
      for (int r = 0; r < 8; ++r) {
        int o = m0 + wm * 64 + mm * 16 + r + hi8;
        int n = n0 + wn * 64 + nn * 16 + lm;
        out[((size_t)b * C_ + o) * W_ + n] = acc[mm][nn][r] + bo[o];
      }
}

// ---------------- host launcher ----------------
extern "C" void kernel_launch(void* const* d_in, const int* in_sizes, int n_in,
                              void* d_out, int out_size, void* d_ws, size_t ws_size,
                              hipStream_t stream) {
  const float* x  = (const float*)d_in[0];
  const float* gw = (const float*)d_in[1];
  const float* gb = (const float*)d_in[2];
  const float* wq = (const float*)d_in[3];
  const float* bq = (const float*)d_in[4];
  const float* wo = (const float*)d_in[5];
  const float* bo = (const float*)d_in[6];
  float* out = (float*)d_out;
  char* ws = (char*)d_ws;

  // workspace layout (bytes, 256-aligned)
  float2* stats = (float2*)(ws + 0);              //  4 KB
  float2* ss    = (float2*)(ws + 4096);           // 64 KB
  u16* wqb   = (u16*)(ws + 69632);                // 1.5 MB
  u16* wob   = (u16*)(ws + 1642496);              // 0.5 MB
  u16* qkv   = (u16*)(ws + 2166784);              // 201 MB
  u16* ctx   = (u16*)(ws + 203493376);            // 1 MB
  u16* w2    = (u16*)(ws + 204541952);            // 8 MB
  float* ctxp = (float*)(ws + 212930560);         // 8 MB (split-K partials)

  k_gn_stats  <<<B_ * GROUPS_, 256, 0, stream>>>(x, stats);
  k_scaleshift<<<(B_ * C_ + 255) / 256, 256, 0, stream>>>(stats, gw, gb, ss);
  k_cvt_w     <<<1024, 256, 0, stream>>>(wq, wo, wqb, wob);
  k_qkv_gemm  <<<dim3(O3_ / 128, W_ / 256, B_), 256, 0, stream>>>(wqb, x, ss, bq, qkv);
  k_softmax   <<<B_ * HID_, 256, 0, stream>>>(qkv);
  k_context   <<<dim3(B_ * HEADS_, 4), 128, 0, stream>>>(qkv, ctxp);
  k_ctx_reduce<<<(B_ * HEADS_ * HD_ * HD_) / 256, 256, 0, stream>>>(ctxp, ctx);
  k_w2        <<<B_ * HEADS_, 256, 0, stream>>>(wob, ctx, w2);
  k_final_gemm<<<dim3(C_ / 128, W_ / 256, B_), 256, 0, stream>>>(w2, qkv, bo, out);
}
